// SimpleNet_55628416418297
// MI455X (gfx1250) — matrix operand order repaired and checked by reading, not compiled, not run on hardware
//
#include <hip/hip_runtime.h>
#include <stdint.h>

// ---------------------------------------------------------------------------
// GraphSAGE on MI455X (gfx1250): bf16 WMMA GEMMs + L2-resident atomic scatter
// ---------------------------------------------------------------------------

typedef __attribute__((ext_vector_type(16))) __bf16 v16bf;
typedef __attribute__((ext_vector_type(8)))  float  v8f;

constexpr int IN_DIM = 128;
constexpr int HID    = 64;
constexpr float LN_EPS = 1e-5f;

__device__ __forceinline__ float4 ld4(const float* p) { return *(const float4*)p; }

__device__ __forceinline__ void cvt8(v16bf& a, int base, float4 lo, float4 hi) {
  a[base + 0] = (__bf16)lo.x; a[base + 1] = (__bf16)lo.y;
  a[base + 2] = (__bf16)lo.z; a[base + 3] = (__bf16)lo.w;
  a[base + 4] = (__bf16)hi.x; a[base + 5] = (__bf16)hi.y;
  a[base + 6] = (__bf16)hi.z; a[base + 7] = (__bf16)hi.w;
}

__device__ __forceinline__ float4 scale4(float4 f, float s) {
  f.x *= s; f.y *= s; f.z *= s; f.w *= s; return f;
}

// A-fragment: 16x32 (MxK) bf16 from row-major fp32 [nrows x ld].
// ISA layout: lanes 0-15 M=0..15 koff=0, lanes 16-31 M=0..15 koff=8.
// VGPR 0-3: K = koff+0..7 ; VGPR 4-7: K = 16+koff+0..7.
template <bool SCALED>
__device__ __forceinline__ v16bf load_afrag(const float* __restrict__ base, int row0,
                                            int nrows, int ld, int k0, int lane,
                                            const float* __restrict__ scale) {
  const int m    = lane & 15;
  const int koff = (lane & 16) ? 8 : 0;
  const int row  = row0 + m;
  v16bf a;
  if (row < nrows) {
    const float* p = base + (size_t)row * ld + (k0 + koff);
    float4 f0 = ld4(p + 0);
    float4 f1 = ld4(p + 4);
    float4 f2 = ld4(p + 16);
    float4 f3 = ld4(p + 20);
    if (SCALED) {
      const float s = scale[row];
      f0 = scale4(f0, s); f1 = scale4(f1, s);
      f2 = scale4(f2, s); f3 = scale4(f3, s);
    }
    cvt8(a, 0, f0, f1);
    cvt8(a, 8, f2, f3);
  } else {
#pragma unroll
    for (int i = 0; i < 16; ++i) a[i] = (__bf16)0.0f;
  }
  return a;
}

// B-fragment: 32x16 (KxN) bf16 from row-major fp32 weights W[N][K] (so B = W^T).
// Layout: lane l holds column N=l&15; kb = 0 (lanes 0-15) or 8 (lanes 16-31);
// elements 0-7: K = kb+0..7 ; elements 8-15: K = 16+kb+0..7 (matches A).
__device__ __forceinline__ v16bf load_bfrag(const float* __restrict__ W, int n0,
                                            int ld, int k0, int lane) {
  const int ncol = n0 + (lane & 15);
  const int kb   = (lane & 16) ? 8 : 0;
  const float* p = W + (size_t)ncol * ld + (k0 + kb);
  float4 f0 = ld4(p + 0);
  float4 f1 = ld4(p + 4);
  float4 f2 = ld4(p + 16);
  float4 f3 = ld4(p + 20);
  v16bf b;
  cvt8(b, 0, f0, f1);
  cvt8(b, 8, f2, f3);
  return b;
}

__device__ __forceinline__ v8f wmma_bf16(v16bf a, v16bf b, v8f c) {
  return __builtin_amdgcn_wmma_f32_16x16x32_bf16(
      /*neg_a=*/false, a, /*neg_b=*/false, b,
      /*c_mod=*/(short)0, c, /*reuse_a=*/false, /*reuse_b=*/false);
}

// ---------------------------------------------------------------------------
// Degree / scatter kernels (memory-bound path; agg is L2-resident)
// ---------------------------------------------------------------------------

__global__ void k_deg(const int* __restrict__ dst, float* __restrict__ deg, int nedges) {
  int e = blockIdx.x * blockDim.x + threadIdx.x;
  if (e < nedges) atomicAdd(&deg[dst[e]], 1.0f);
}

__global__ void k_invdeg(float* __restrict__ deg, int n) {
  int i = blockIdx.x * blockDim.x + threadIdx.x;
  if (i < n) deg[i] = 1.0f / fmaxf(deg[i], 1.0f);
}

// 16 threads per edge; each handles 4 consecutive features (float4 gather).
__global__ void k_scatter(const int* __restrict__ src, const int* __restrict__ dst,
                          const float* __restrict__ h, float* __restrict__ agg,
                          int nedges) {
  int t = blockIdx.x * blockDim.x + threadIdx.x;
  int e = t >> 4;
  if (e >= nedges) return;
  int c = (t & 15) << 2;
  int s = src[e];
  int d = dst[e];
  float4 v = *(const float4*)(h + (size_t)s * HID + c);
  float* p = agg + (size_t)d * HID + c;
  atomicAdd(p + 0, v.x);
  atomicAdd(p + 1, v.y);
  atomicAdd(p + 2, v.z);
  atomicAdd(p + 3, v.w);
}

// ---------------------------------------------------------------------------
// FC projection: h = relu(x @ fc_W^T + fc_b), x[n,128], fc_W[64,128]
// One wave per 16 rows; 4 N-tiles x 4 K-steps = 16 WMMAs.
// ---------------------------------------------------------------------------

__global__ void __launch_bounds__(256)
k_fc(const float* __restrict__ x, const float* __restrict__ W,
     const float* __restrict__ bias, float* __restrict__ h, int n) {
  const int gwave = (blockIdx.x * blockDim.x + threadIdx.x) >> 5;
  const int lane  = threadIdx.x & 31;
  const int row0  = gwave * 16;
  if (row0 >= n) return;  // uniform per wave -> EXEC stays all-ones for WMMA

  v8f acc[4] = {};
#pragma unroll
  for (int kk = 0; kk < IN_DIM; kk += 32) {
    v16bf a = load_afrag<false>(x, row0, n, IN_DIM, kk, lane, nullptr);
#pragma unroll
    for (int j = 0; j < 4; ++j) {
      v16bf b = load_bfrag(W, 16 * j, IN_DIM, kk, lane);
      acc[j] = wmma_bf16(a, b, acc[j]);
    }
  }

  // C layout: VGPR r -> row (r | r+8), col = lane&15 (+16*j per N-tile)
  const int col   = lane & 15;
  const int rbase = (lane >> 4) << 3;
  float bv[4];
#pragma unroll
  for (int j = 0; j < 4; ++j) bv[j] = bias[col + 16 * j];
#pragma unroll
  for (int r = 0; r < 8; ++r) {
    const int row = row0 + rbase + r;
    if (row < n) {
#pragma unroll
      for (int j = 0; j < 4; ++j) {
        float v = acc[j][r] + bv[j];
        h[(size_t)row * HID + col + 16 * j] = fmaxf(v, 0.0f);
      }
    }
  }
}

// ---------------------------------------------------------------------------
// SAGE layer: out = (agg*inv_deg) @ Wl^T + h @ Wr^T + b  [+ LN + ReLU]
// LN over 64 features: each row spans 16 lanes x 4 acc tiles -> xor-shuffle
// reduction with masks 1/2/4/8 (confined to each 16-lane half on wave32).
// ---------------------------------------------------------------------------

template <bool LN>
__global__ void __launch_bounds__(256)
k_sage(const float* __restrict__ agg, const float* __restrict__ invdeg,
       const float* __restrict__ h, const float* __restrict__ Wl,
       const float* __restrict__ Wr, const float* __restrict__ bias,
       const float* __restrict__ g, const float* __restrict__ bb,
       float* __restrict__ out, int n) {
  const int gwave = (blockIdx.x * blockDim.x + threadIdx.x) >> 5;
  const int lane  = threadIdx.x & 31;
  const int row0  = gwave * 16;
  if (row0 >= n) return;

  v8f acc[4] = {};
#pragma unroll
  for (int kk = 0; kk < HID; kk += 32) {
    v16bf a_agg = load_afrag<true >(agg, row0, n, HID, kk, lane, invdeg);
    v16bf a_h   = load_afrag<false>(h,   row0, n, HID, kk, lane, nullptr);
#pragma unroll
    for (int j = 0; j < 4; ++j) {
      v16bf b_l = load_bfrag(Wl, 16 * j, HID, kk, lane);
      acc[j] = wmma_bf16(a_agg, b_l, acc[j]);
      v16bf b_r = load_bfrag(Wr, 16 * j, HID, kk, lane);
      acc[j] = wmma_bf16(a_h, b_r, acc[j]);
    }
  }

  const int col   = lane & 15;
  const int rbase = (lane >> 4) << 3;
  float bv[4], gv[4], b2v[4];
#pragma unroll
  for (int j = 0; j < 4; ++j) {
    bv[j] = bias[col + 16 * j];
    if (LN) {
      gv[j]  = g[col + 16 * j];
      b2v[j] = bb[col + 16 * j];
    }
  }

#pragma unroll
  for (int r = 0; r < 8; ++r) {
    float v[4];
#pragma unroll
    for (int j = 0; j < 4; ++j) v[j] = acc[j][r] + bv[j];
    if (LN) {
      float s = v[0] + v[1] + v[2] + v[3];
      s += __shfl_xor(s, 1); s += __shfl_xor(s, 2);
      s += __shfl_xor(s, 4); s += __shfl_xor(s, 8);
      const float mu = s * (1.0f / 64.0f);
      float q = 0.0f;
#pragma unroll
      for (int j = 0; j < 4; ++j) { float d = v[j] - mu; q += d * d; }
      q += __shfl_xor(q, 1); q += __shfl_xor(q, 2);
      q += __shfl_xor(q, 4); q += __shfl_xor(q, 8);
      const float rstd = rsqrtf(q * (1.0f / 64.0f) + LN_EPS);
#pragma unroll
      for (int j = 0; j < 4; ++j)
        v[j] = fmaxf(gv[j] * (v[j] - mu) * rstd + b2v[j], 0.0f);
    }
    const int row = row0 + rbase + r;
    if (row < n) {
#pragma unroll
      for (int j = 0; j < 4; ++j)
        out[(size_t)row * HID + col + 16 * j] = v[j];
    }
  }
}

// ---------------------------------------------------------------------------
// Host driver
// ---------------------------------------------------------------------------

extern "C" void kernel_launch(void* const* d_in, const int* in_sizes, int n_in,
                              void* d_out, int out_size, void* d_ws, size_t ws_size,
                              hipStream_t stream) {
  const float* x     = (const float*)d_in[0];
  const int*   edge  = (const int*)  d_in[1];
  // d_in[2] = batch (unused by the reference output)
  const float* fc_W  = (const float*)d_in[3];
  const float* fc_b  = (const float*)d_in[4];
  const float* lin_l = (const float*)d_in[5];
  const float* lin_r = (const float*)d_in[6];
  const float* lin_b = (const float*)d_in[7];
  const float* ln_g  = (const float*)d_in[8];
  const float* ln_b  = (const float*)d_in[9];
  float* out = (float*)d_out;

  const int n      = in_sizes[0] / IN_DIM;
  const int nedges = in_sizes[1] / 2;
  const int* src = edge;           // edge_index[0]
  const int* dst = edge + nedges;  // edge_index[1]

  // workspace layout (floats): deg | h0 | h1 | agg
  float* ws   = (float*)d_ws;
  size_t n_al = (size_t)((n + 3) & ~3);
  float* deg  = ws;
  float* h0   = deg + n_al;
  float* h1   = h0 + (size_t)n * HID;
  float* agg  = h1 + (size_t)n * HID;

  // 1) degrees -> inv_deg (in place)
  hipMemsetAsync(deg, 0, (size_t)n * sizeof(float), stream);
  k_deg<<<(nedges + 255) / 256, 256, 0, stream>>>(dst, deg, nedges);
  k_invdeg<<<(n + 255) / 256, 256, 0, stream>>>(deg, n);

  // 2) FC projection + ReLU
  const int nwaves = (n + 15) / 16;
  const int gblk   = (nwaves * 32 + 255) / 256;
  k_fc<<<gblk, 256, 0, stream>>>(x, fc_W, fc_b, h0, n);

  // 3) three SAGE layers
  const long sthreads = (long)nedges * 16;
  const int  sblk     = (int)((sthreads + 255) / 256);
  float* hc = h0;
  float* hn = h1;
  for (int layer = 0; layer < 3; ++layer) {
    hipMemsetAsync(agg, 0, (size_t)n * HID * sizeof(float), stream);
    k_scatter<<<sblk, 256, 0, stream>>>(src, dst, hc, agg, nedges);
    const float* Wl = lin_l + (size_t)layer * HID * HID;
    const float* Wr = lin_r + (size_t)layer * HID * HID;
    const float* bi = lin_b + (size_t)layer * HID;
    float* o = (layer == 2) ? out : hn;
    if (layer < 2) {
      k_sage<true><<<gblk, 256, 0, stream>>>(agg, deg, hc, Wl, Wr, bi,
                                             ln_g + (size_t)layer * HID,
                                             ln_b + (size_t)layer * HID, o, n);
    } else {
      k_sage<false><<<gblk, 256, 0, stream>>>(agg, deg, hc, Wl, Wr, bi,
                                              nullptr, nullptr, o, n);
    }
    float* t = hc; hc = hn; hn = t;
  }
}
